// DeformableAttnBlocks_2860448219493
// MI455X (gfx1250) — compile-verified, matrix operand order checked by reading, not run on hardware
//
#include <hip/hip_runtime.h>
#include <cstdint>

typedef __attribute__((ext_vector_type(16))) _Float16 v16h;
typedef __attribute__((ext_vector_type(8)))  float    v8f;
typedef unsigned int u32x4 __attribute__((ext_vector_type(4)));
typedef int          i32x4 __attribute__((ext_vector_type(4)));
typedef int          i32x8 __attribute__((ext_vector_type(8)));

// ---------------- problem constants ----------------
static constexpr int HWPIX = 64 * 64;
static constexpr int C = 64, T = 3, M = 8, L = 3, P = 12, KK = 9;
static constexpr int CIN = T * C + 4;          // 196
static constexpr int CINP = 224;               // padded to multiple of 32
static constexpr int K33 = 9 * 64;             // 576 (im2col K, k = tap*64 + c)
static constexpr int OC_AW = M * L * P * KK;   // 2592
static constexpr int OC_SO = M * L * P * 2;    // 576
static constexpr int OC_PO = M * L * P;        // 288
static constexpr int CV = T * C;               // 192

#if __has_builtin(__builtin_amdgcn_tensor_load_to_lds)
#define HAVE_TDM 1
#else
#define HAVE_TDM 0
#endif

__device__ __forceinline__ uint32_t lds_off(const void* p) {
  return (uint32_t)(uintptr_t)p;  // LDS aperture: addr[31:0] = LDS byte offset
}

// TDM 2D tile load: tile_h rows of tile_w f16 elements, row stride
// `row_stride` elements, from `gaddr` (tile start) into LDS at lds_byte_off.
// OOB rows (>= tensor_h) are zero-filled by the TDM.
__device__ __forceinline__ void tdm_load_2d(uint32_t lds_byte_off,
                                            const void* gaddr, uint32_t tile_w,
                                            uint32_t tile_h, uint32_t tensor_w,
                                            uint32_t tensor_h,
                                            uint64_t row_stride) {
#if HAVE_TDM
  const uint64_t ga = (uint64_t)(uintptr_t)gaddr;
  u32x4 g0;
  g0[0] = 1u;  // count=1 (valid), user mode, no gather
  g0[1] = lds_byte_off;
  g0[2] = (uint32_t)ga;
  g0[3] = (uint32_t)((ga >> 32) & 0x1FFFFFFull) | (2u << 30);  // type=2
  i32x8 g1;
  g1[0] = (int)(1u << 16);  // data_size=1 -> 2 bytes; no multicast/flags
  g1[1] = (int)((tensor_w & 0xFFFFu) << 16);                       // dim0 lo
  g1[2] = (int)(((tensor_w >> 16) & 0xFFFFu) | ((tensor_h & 0xFFFFu) << 16));
  g1[3] = (int)(((tensor_h >> 16) & 0xFFFFu) | ((tile_w & 0xFFFFu) << 16));
  g1[4] = (int)(tile_h & 0xFFFFu);  // tile_dim1; tile_dim2 = 0
  g1[5] = (int)(uint32_t)row_stride;
  g1[6] = (int)((row_stride >> 32) & 0xFFFFull);  // dim1_stride = 0
  g1[7] = 0;
  i32x4 z4 = {0, 0, 0, 0};
#if defined(__clang_major__) && (__clang_major__ >= 23)
  i32x8 z8 = {0, 0, 0, 0, 0, 0, 0, 0};
  __builtin_amdgcn_tensor_load_to_lds(g0, g1, z4, z4, z8, 0);
#else
  __builtin_amdgcn_tensor_load_to_lds(g0, g1, z4, z4, 0);
#endif
#endif
}

template <int N>
__device__ __forceinline__ void tdm_wait() {
#if HAVE_TDM
#if __has_builtin(__builtin_amdgcn_s_wait_tensorcnt)
  __builtin_amdgcn_s_wait_tensorcnt((short)N);  // immediate required
#endif
#endif
}

// =====================================================================
// Dense WMMA GEMM with TDM double-buffered LDS staging.
//   Out[pix][oc] = act( W[oc][K] * Xt[pix][K] + bias[oc] )
//   Block: 8 waves = 2 oc-tiles x 4 pixel-tiles (32 oc x 64 pix).
//   Per K-step (32): A tile 32x32, B tile 64x32 staged to LDS by TDM.
//   Fragment LDS reads are 16B groups -> ds_load_b128.
// =====================================================================
__device__ __forceinline__ void dense_stage(const _Float16* Abase,
                                            const _Float16* Bbase, _Float16* As,
                                            _Float16* Bs, int K, int k0,
                                            int tid) {
#if HAVE_TDM
  if (tid < 32) {
    tdm_load_2d(lds_off(As), Abase + k0, 32, 32, 32, 32, (uint64_t)K);
    tdm_load_2d(lds_off(Bs), Bbase + k0, 32, 64, 32, 64, (uint64_t)K);
  }
#else
  for (int i = tid; i < 32 * 16; i += 256) {
    int r = i >> 4, cp = (i & 15) << 1;
    *(uint32_t*)(As + r * 32 + cp) =
        *(const uint32_t*)(Abase + (size_t)r * K + k0 + cp);
  }
  for (int i = tid; i < 64 * 16; i += 256) {
    int r = i >> 4, cp = (i & 15) << 1;
    *(uint32_t*)(Bs + r * 32 + cp) =
        *(const uint32_t*)(Bbase + (size_t)r * K + k0 + cp);
  }
#endif
}

template <bool LRELU, bool OUT_F16>
__global__ __launch_bounds__(256) void gemm_dense_wmma_kernel(
    const _Float16* __restrict__ Wm, const _Float16* __restrict__ X,
    const float* __restrict__ bias, void* __restrict__ out, int OC, int K) {
  __shared__ _Float16 As[2][32 * 32];
  __shared__ _Float16 Bs[2][64 * 32];
  const int tid = threadIdx.x;
  const int lane = tid & 31, wave = tid >> 5;
  const int m0 = blockIdx.y * 32;
  const int n0 = blockIdx.x * 64;
  const int half = lane >> 4, lid = lane & 15;
  const int arow = (wave >> 2) * 16 + lid;  // local weight row
  const int brow = (wave & 3) * 16 + lid;   // local pixel row
  const int col = n0 + brow;
  const int ocb = m0 + (wave >> 2) * 16 + (half << 3);

  v8f acc;
#pragma unroll
  for (int r = 0; r < 8; ++r) acc[r] = bias[ocb + r];

  const _Float16* Abase = Wm + (size_t)m0 * K;
  const _Float16* Bbase = X + (size_t)n0 * K;
  const int nk = K >> 5;

  dense_stage(Abase, Bbase, As[0], Bs[0], K, 0, tid);
  for (int i = 0; i < nk; ++i) {
    if (i + 1 < nk) {
      dense_stage(Abase, Bbase, As[(i + 1) & 1], Bs[(i + 1) & 1], K,
                  (i + 1) << 5, tid);
      tdm_wait<2>();  // current stage done; next stage's 2 ops in flight
    } else {
      tdm_wait<0>();  // drain
    }
    __syncthreads();
    const _Float16* Ab = As[i & 1];
    const _Float16* Bb = Bs[i & 1];
    union { v16h v; u32x4 q[2]; } a, b;
    a.q[0] = *(const u32x4*)(Ab + arow * 32 + (half << 3));
    a.q[1] = *(const u32x4*)(Ab + arow * 32 + 16 + (half << 3));
    b.q[0] = *(const u32x4*)(Bb + brow * 32 + (half << 3));
    b.q[1] = *(const u32x4*)(Bb + brow * 32 + 16 + (half << 3));
    acc = __builtin_amdgcn_wmma_f32_16x16x32_f16(false, a.v, false, b.v,
                                                 (short)0, acc, false, false);
    __syncthreads();
  }

#pragma unroll
  for (int r = 0; r < 8; ++r) {
    float y = acc[r];
    if (LRELU) y = (y >= 0.f) ? y : 0.1f * y;
    if (OUT_F16)
      ((_Float16*)out)[(size_t)col * OC + ocb + r] = (_Float16)y;
    else
      ((float*)out)[(size_t)col * OC + ocb + r] = y;
  }
}

// =====================================================================
// 3x3 conv (pad=1) as implicit GEMM. One image row (64 pixels) per block.
// Entire receptive field (3 input rows, 24KB) + 32x576 weight slab (36KB)
// TDM-staged into LDS once; K-loop is pure ds_load_b128 + WMMA.
// =====================================================================
__global__ __launch_bounds__(256) void conv3_wmma_kernel(
    const _Float16* __restrict__ Wm, const _Float16* __restrict__ X,
    const float* __restrict__ bias, _Float16* __restrict__ out) {
  __shared__ _Float16 Aslab[32 * K33];  // 36 KB
  __shared__ _Float16 Xs[192 * 64];     // 24 KB: 3 image rows, NHWC
  const int tid = threadIdx.x;
  const int lane = tid & 31, wave = tid >> 5;
  const int m0 = blockIdx.y * 32;
  const int py = blockIdx.x;
  const int ystart = (py == 0) ? 0 : (py - 1);
  const int half = lane >> 4, lid = lane & 15;
  const int arow = (wave >> 2) * 16 + lid;
  const int px = (wave & 3) * 16 + lid;
  const int ocb = m0 + (wave >> 2) * 16 + (half << 3);

#if HAVE_TDM
  if (tid < 32) {
    tdm_load_2d(lds_off(Aslab), Wm + (size_t)m0 * K33, K33, 32, K33, 32, K33);
    tdm_load_2d(lds_off(Xs), X + (size_t)ystart * 64 * 64, 64, 192, 64,
                (uint32_t)(HWPIX - ystart * 64), 64);  // OOB rows -> zeros
  }
  tdm_wait<0>();
#else
  for (int i = tid; i < 32 * (K33 / 2); i += 256) {
    int r = i / (K33 / 2), cp = (i % (K33 / 2)) * 2;
    *(uint32_t*)(Aslab + r * K33 + cp) =
        *(const uint32_t*)(Wm + (size_t)(m0 + r) * K33 + cp);
  }
  for (int i = tid; i < 192 * 32; i += 256) {
    int r = i >> 5, cp = (i & 31) << 1;
    int gp = ystart * 64 + r;
    uint32_t v = 0;
    if (gp < HWPIX) v = *(const uint32_t*)(X + (size_t)gp * 64 + cp);
    *(uint32_t*)(Xs + r * 64 + cp) = v;
  }
#endif
  __syncthreads();

  v8f acc;
#pragma unroll
  for (int r = 0; r < 8; ++r) acc[r] = bias[ocb + r];

  for (int k0 = 0; k0 < K33; k0 += 32) {
    union { v16h v; u32x4 q[2]; } a, b;
    a.q[0] = *(const u32x4*)(Aslab + arow * K33 + k0 + (half << 3));
    a.q[1] = *(const u32x4*)(Aslab + arow * K33 + k0 + 16 + (half << 3));
#pragma unroll
    for (int g = 0; g < 2; ++g) {  // each 16B group stays within one tap
      const int kg = k0 + g * 16 + (half << 3);
      const int tap = kg >> 6, c = kg & 63;
      const int sy = py + tap / 3 - 1;
      const int sx = px + tap % 3 - 1;
      u32x4 bv = {0u, 0u, 0u, 0u};
      if (sx >= 0 && sx < 64 && sy >= 0 && sy < 64)
        bv = *(const u32x4*)(Xs + ((sy - ystart) * 64 + sx) * 64 + c);
      b.q[g] = bv;
    }
    acc = __builtin_amdgcn_wmma_f32_16x16x32_f16(false, a.v, false, b.v,
                                                 (short)0, acc, false, false);
  }

  const int colq = py * 64 + px;
#pragma unroll
  for (int r = 0; r < 8; ++r) {
    float y = acc[r];
    y = (y >= 0.f) ? y : 0.1f * y;  // lrelu
    out[(size_t)colq * 64 + ocb + r] = (_Float16)y;
  }
}

// ---------------- weight repack (f32 -> f16 GEMM layouts) ----------------
__global__ void pack_1x1_kernel(const float* __restrict__ w,
                                _Float16* __restrict__ out, int OC, int Cin,
                                int Kpad) {
  int i = blockIdx.x * 256 + threadIdx.x;
  if (i >= OC * Kpad) return;
  int oc = i / Kpad, k = i - oc * Kpad;
  out[i] = (_Float16)(k < Cin ? w[(size_t)oc * Cin + k] : 0.f);
}

__global__ void pack_3x3_kernel(const float* __restrict__ w,
                                _Float16* __restrict__ out, int OC) {
  int i = blockIdx.x * 256 + threadIdx.x;
  if (i >= OC * K33) return;
  int oc = i / K33, k = i - oc * K33;
  int tap = k >> 6, c = k & 63, ky = tap / 3, kx = tap % 3;
  out[i] = (_Float16)w[(((size_t)oc * 64 + c) * 3 + ky) * 3 + kx];
}

// ---------------- flow warp + input staging ----------------
struct Bilin {
  int idx[4];
  float w[4];
  __device__ void init(float ix, float iy) {
    float fx = floorf(ix), fy = floorf(iy);
    int x0 = (int)fx, y0 = (int)fy;
    float wx = ix - fx, wy = iy - fy;
#pragma unroll
    for (int t = 0; t < 4; ++t) {
      int xx = x0 + (t & 1), yy = y0 + (t >> 1);
      bool ok = (xx >= 0) && (xx < 64) && (yy >= 0) && (yy < 64);
      idx[t] = ok ? (yy * 64 + xx) : -1;
      w[t] = ((t & 1) ? wx : 1.f - wx) * ((t >> 1) ? wy : 1.f - wy);
    }
  }
  __device__ float sample(const float* ch) const {
    float a = 0.f;
#pragma unroll
    for (int t = 0; t < 4; ++t)
      if (idx[t] >= 0) a += w[t] * ch[idx[t]];
    return a;
  }
};

__global__ __launch_bounds__(256) void prep_kernel(
    const float* __restrict__ x, const float* __restrict__ flow1,
    const float* __restrict__ flow2, _Float16* __restrict__ extra,
    _Float16* __restrict__ xv, float* __restrict__ flowc) {
  int q = blockIdx.x * 256 + threadIdx.x;
  if (q >= HWPIX) return;
  int px = q & 63, py = q >> 6;
  float f1x = flow1[q], f1y = flow1[HWPIX + q];
  Bilin b1;
  b1.init((float)px + f1x, (float)py + f1y);
  float f2x = f1x + b1.sample(flow2);
  float f2y = f1y + b1.sample(flow2 + HWPIX);
  Bilin b2;
  b2.init((float)px + f2x, (float)py + f2y);

  _Float16* e = extra + (size_t)q * CINP;
  _Float16* v = xv + (size_t)q * CV;
#pragma unroll 4
  for (int c = 0; c < 64; ++c) {
    float x0 = x[(size_t)c * HWPIX + q];
    float w1 = b1.sample(x + (size_t)(64 + c) * HWPIX);
    float w2 = b2.sample(x + (size_t)(128 + c) * HWPIX);
    e[c] = (_Float16)x0;
    e[64 + c] = (_Float16)w1;
    e[128 + c] = (_Float16)w2;
    v[c] = (_Float16)x0;
    v[64 + c] = (_Float16)x[(size_t)(64 + c) * HWPIX + q];
    v[128 + c] = (_Float16)x[(size_t)(128 + c) * HWPIX + q];
  }
  e[192] = (_Float16)f1x;
  e[193] = (_Float16)f1y;
  e[194] = (_Float16)f2x;
  e[195] = (_Float16)f2y;
#pragma unroll
  for (int c = CIN; c < CINP; ++c) e[c] = (_Float16)0.f;
  flowc[q * 4 + 0] = f1x;
  flowc[q * 4 + 1] = f1y;
  flowc[q * 4 + 2] = f2x;
  flowc[q * 4 + 3] = f2y;
}

// ---------------- deformable attention core (fused softmax/sigmoid) ------
__global__ __launch_bounds__(256) void msdeform_kernel(
    const float* __restrict__ value,   // [4096][192] (ch = l*64+m*8+d)
    const float* __restrict__ so_raw,  // [4096][576]
    const float* __restrict__ aw_raw,  // [4096][2592]
    const float* __restrict__ po_raw,  // [4096][288]
    const float* __restrict__ flowc,   // [4096][4]
    _Float16* __restrict__ out) {      // [4096][64] (ch = m*8+d)
  int tid = blockIdx.x * 256 + threadIdx.x;  // tid = q*8 + m
  if (tid >= HWPIX * M) return;
  int q = tid >> 3, m = tid & 7;
  int px = q & 63, py = q >> 6;

  const float* aw = aw_raw + (size_t)q * OC_AW + m * (L * P * KK);
  __builtin_prefetch(aw, 0, 1);
  float mx = -1e30f;
  for (int j = 0; j < L * P * KK; ++j) mx = fmaxf(mx, aw[j]);
  float se = 0.f;
  for (int j = 0; j < L * P * KK; ++j) se += __expf(aw[j] - mx);
  float inv = 1.f / se;

  const float* so = so_raw + (size_t)q * OC_SO;
  const float* po = po_raw + (size_t)q * OC_PO;
  float refx = ((float)px + 0.5f) / 64.f;
  float refy = ((float)py + 0.5f) / 64.f;

  float acc[8];
#pragma unroll
  for (int d = 0; d < 8; ++d) acc[d] = 0.f;

  for (int l = 0; l < L; ++l) {
    float fx = (l == 0) ? 0.f : flowc[q * 4 + (l - 1) * 2 + 0];
    float fy = (l == 0) ? 0.f : flowc[q * 4 + (l - 1) * 2 + 1];
    const int cb = l * 64 + m * 8;
    for (int p = 0; p < P; ++p) {
      int ch = (m * L + l) * P + p;
      float ox = so[ch * 2 + 0] + fx;
      float oy = so[ch * 2 + 1] + fy;
      float pov = 3.f / (1.f + __expf(-po[ch]));  // sigmoid * 3
      for (int kk = 0; kk < KK; ++kk) {
        float pnx = (float)(kk / 3 - 1), pny = (float)(kk % 3 - 1);
        float tx = ox + pnx * pov, ty = oy + pny * pov;
        float ix = (refx + tx / 64.f) * 64.f - 0.5f;  // align_corners=False
        float iy = (refy + ty / 64.f) * 64.f - 0.5f;
        float wgt = __expf(aw[l * (P * KK) + p * KK + kk] - mx) * inv;
        float fx0 = floorf(ix), fy0 = floorf(iy);
        int x0 = (int)fx0, y0 = (int)fy0;
        float wx = ix - fx0, wy = iy - fy0;
#pragma unroll
        for (int t = 0; t < 4; ++t) {
          int xx = x0 + (t & 1), yy = y0 + (t >> 1);
          if (xx < 0 || xx > 63 || yy < 0 || yy > 63) continue;
          float tw =
              wgt * ((t & 1) ? wx : 1.f - wx) * ((t >> 1) ? wy : 1.f - wy);
          const float* base = value + (size_t)(yy * 64 + xx) * CV + cb;
#pragma unroll
          for (int d = 0; d < 8; ++d) acc[d] += tw * base[d];
        }
      }
    }
  }
#pragma unroll
  for (int d = 0; d < 8; ++d)
    out[(size_t)q * C + m * 8 + d] = (_Float16)acc[d];
}

// ---------------- residual epilogue (NHWC f32 -> NCHW + x[:,0]) ----------
__global__ void residual_kernel(const float* __restrict__ op2,
                                const float* __restrict__ x,
                                float* __restrict__ out) {
  int i = blockIdx.x * 256 + threadIdx.x;
  if (i >= C * HWPIX) return;
  int c = i >> 12, q = i & (HWPIX - 1);
  out[i] = op2[(size_t)q * C + c] + x[(size_t)c * HWPIX + q];
}

// =====================================================================
extern "C" void kernel_launch(void* const* d_in, const int* in_sizes, int n_in,
                              void* d_out, int out_size, void* d_ws,
                              size_t ws_size, hipStream_t stream) {
  const float* x = (const float*)d_in[0];
  const float* flow1 = (const float*)d_in[1];
  const float* flow2 = (const float*)d_in[2];
  const float* tw[3][4], *tb[3][4];  // trunks: 0=aw, 1=so, 2=po
  for (int t = 0; t < 3; ++t)
    for (int lyr = 0; lyr < 4; ++lyr) {
      tw[t][lyr] = (const float*)d_in[5 + t * 8 + lyr * 2];
      tb[t][lyr] = (const float*)d_in[5 + t * 8 + lyr * 2 + 1];
    }
  const float* vp_w = (const float*)d_in[29];
  const float* vp_b = (const float*)d_in[30];
  const float* op_w1 = (const float*)d_in[31];
  const float* op_b1 = (const float*)d_in[32];
  const float* op_w2 = (const float*)d_in[33];
  const float* op_b2 = (const float*)d_in[34];
  const int trunk_oc[3] = {OC_AW, OC_SO, OC_PO};

  // -------- workspace carve-up --------
  char* ws = (char*)d_ws;
  size_t off = 0;
  auto take = [&](size_t elems, size_t esz) {
    void* p = ws + off;
    off += (elems * esz + 255) & ~(size_t)255;
    return p;
  };
  _Float16* wT1[3], *wT2[3], *wT3[3], *wT4[3];
  for (int t = 0; t < 3; ++t) {
    wT1[t] = (_Float16*)take(64 * CINP, 2);
    wT2[t] = (_Float16*)take(64 * K33, 2);
    wT3[t] = (_Float16*)take(64 * K33, 2);
    wT4[t] = (_Float16*)take((size_t)trunk_oc[t] * 64, 2);
  }
  _Float16* wVP = (_Float16*)take(CV * CV, 2);
  _Float16* wO1 = (_Float16*)take(64 * 64, 2);
  _Float16* wO2 = (_Float16*)take(64 * 64, 2);
  _Float16* extra = (_Float16*)take((size_t)HWPIX * CINP, 2);
  _Float16* xv = (_Float16*)take((size_t)HWPIX * CV, 2);
  _Float16* hA = (_Float16*)take((size_t)HWPIX * 64, 2);
  _Float16* hB = (_Float16*)take((size_t)HWPIX * 64, 2);
  _Float16* attn = (_Float16*)take((size_t)HWPIX * 64, 2);
  _Float16* op1 = (_Float16*)take((size_t)HWPIX * 64, 2);
  float* flowc = (float*)take((size_t)HWPIX * 4, 4);
  float* value = (float*)take((size_t)HWPIX * CV, 4);
  float* so_raw = (float*)take((size_t)HWPIX * OC_SO, 4);
  float* po_raw = (float*)take((size_t)HWPIX * OC_PO, 4);
  float* aw_raw = (float*)take((size_t)HWPIX * OC_AW, 4);
  float* op2 = (float*)take((size_t)HWPIX * 64, 4);

  auto blocks = [](int n) { return (n + 255) / 256; };

  // -------- weight repack --------
  for (int t = 0; t < 3; ++t) {
    pack_1x1_kernel<<<blocks(64 * CINP), 256, 0, stream>>>(tw[t][0], wT1[t], 64,
                                                           CIN, CINP);
    pack_3x3_kernel<<<blocks(64 * K33), 256, 0, stream>>>(tw[t][1], wT2[t], 64);
    pack_3x3_kernel<<<blocks(64 * K33), 256, 0, stream>>>(tw[t][2], wT3[t], 64);
    pack_1x1_kernel<<<blocks(trunk_oc[t] * 64), 256, 0, stream>>>(
        tw[t][3], wT4[t], trunk_oc[t], 64, 64);
  }
  pack_1x1_kernel<<<blocks(CV * CV), 256, 0, stream>>>(vp_w, wVP, CV, CV, CV);
  pack_1x1_kernel<<<blocks(64 * 64), 256, 0, stream>>>(op_w1, wO1, 64, 64, 64);
  pack_1x1_kernel<<<blocks(64 * 64), 256, 0, stream>>>(op_w2, wO2, 64, 64, 64);

  // -------- flow warp + NHWC f16 staging --------
  prep_kernel<<<blocks(HWPIX), 256, 0, stream>>>(x, flow1, flow2, extra, xv,
                                                 flowc);

  // -------- value projection (192x192 GEMM, f32 out) --------
  gemm_dense_wmma_kernel<false, false><<<dim3(64, CV / 32), 256, 0, stream>>>(
      wVP, xv, vp_b, value, CV, CV);

  // -------- three conv trunks --------
  float* traw[3] = {aw_raw, so_raw, po_raw};
  for (int t = 0; t < 3; ++t) {
    gemm_dense_wmma_kernel<true, true><<<dim3(64, 2), 256, 0, stream>>>(
        wT1[t], extra, tb[t][0], hA, 64, CINP);
    conv3_wmma_kernel<<<dim3(64, 2), 256, 0, stream>>>(wT2[t], hA, tb[t][1],
                                                       hB);
    conv3_wmma_kernel<<<dim3(64, 2), 256, 0, stream>>>(wT3[t], hB, tb[t][2],
                                                       hA);
    gemm_dense_wmma_kernel<false, false>
        <<<dim3(64, trunk_oc[t] / 32), 256, 0, stream>>>(
            wT4[t], hA, tb[t][3], traw[t], trunk_oc[t], 64);
  }

  // -------- deformable attention (fused softmax + sigmoid + sampling) ----
  msdeform_kernel<<<blocks(HWPIX * M), 256, 0, stream>>>(value, so_raw, aw_raw,
                                                         po_raw, flowc, attn);

  // -------- output projection + residual --------
  gemm_dense_wmma_kernel<true, true><<<dim3(64, 2), 256, 0, stream>>>(
      wO1, attn, op_b1, op1, 64, 64);
  gemm_dense_wmma_kernel<false, false><<<dim3(64, 2), 256, 0, stream>>>(
      wO2, op1, op_b2, op2, 64, 64);
  residual_kernel<<<blocks(C * HWPIX), 256, 0, stream>>>(op2, x,
                                                         (float*)d_out);
}
